// TransformerEncoder_41162966565304
// MI455X (gfx1250) — matrix operand order repaired and checked by reading, not compiled, not run on hardware
//
#include <hip/hip_runtime.h>
#include <hip/hip_bf16.h>
#include <math.h>

typedef __bf16 bf16_t;
typedef __attribute__((ext_vector_type(16))) __bf16 v16bf;
typedef __attribute__((ext_vector_type(8)))  __bf16 v8bf;
typedef __attribute__((ext_vector_type(8)))  float  v8f;

#define B_   16
#define S_   1024
#define D_   768
#define H_   12
#define HD_  64
#define MLP_ 3072
#define NT_  (B_ * S_)   // 16384 tokens

// ---------------------------------------------------------------------------
// WMMA helpers (CDNA5 gfx1250, wave32)
// ---------------------------------------------------------------------------
__device__ __forceinline__ v8f wmma_bf16(v16bf a, v16bf b, v8f c) {
  // D = A(16x32) * B(32x16) + C, f32 accumulate
  return __builtin_amdgcn_wmma_f32_16x16x32_bf16(false, a, false, b, (short)0, c,
                                                 false, false);
}

// A-fragment: 16x32 bf16 matrix, row-major source with element stride `stride`.
// Layout: lanes 0-15 row M=lane, K=[0..15];
//         lanes 16-31 row M=lane-16, K=[16..31].
__device__ __forceinline__ v16bf load_a_frag(const bf16_t* base, int stride, int lane) {
  const int m  = lane & 15;
  const int kb = (lane < 16) ? 0 : 16;
  const bf16_t* p = base + (size_t)m * stride + kb;
  union { v16bf v; v8bf h[2]; } u;
  u.h[0] = *(const v8bf*)(p);        // K = kb + [0..7]
  u.h[1] = *(const v8bf*)(p + 8);    // K = kb + [8..15]
  return u.v;
}

// B-fragment: 32x16, element (k,n) = src[n*stride + k] (src row-major [N][K]).
// Layout used here (same as A): lanes 0-15 col N=lane, K=[0..15]; lanes 16-31 col N=lane-16, K=[16..31].
__device__ __forceinline__ v16bf load_b_frag(const bf16_t* base, int stride, int lane) {
  const int n  = lane & 15;
  const int kb = (lane < 16) ? 0 : 16;
  return *(const v16bf*)(base + (size_t)n * stride + kb);
}

// Async global -> LDS staging (tracked by ASYNCcnt).
// lds_off: wave-relative LDS byte offset (low 32 bits of the generic pointer);
// gptr: per-lane 64-bit global address; each lane moves 16 bytes.
__device__ __forceinline__ void async_load_b128(unsigned lds_off, const void* gptr) {
  asm volatile("global_load_async_to_lds_b128 %0, %1, off"
               :: "v"(lds_off), "v"(gptr)
               : "memory");
}

// Reductions across one 16-lane half (rows of the C/D layout live per half).
__device__ __forceinline__ float half_max(float v) {
  v = fmaxf(v, __shfl_xor(v, 1, 32));
  v = fmaxf(v, __shfl_xor(v, 2, 32));
  v = fmaxf(v, __shfl_xor(v, 4, 32));
  v = fmaxf(v, __shfl_xor(v, 8, 32));
  return v;
}
__device__ __forceinline__ float half_sum(float v) {
  v += __shfl_xor(v, 1, 32);
  v += __shfl_xor(v, 2, 32);
  v += __shfl_xor(v, 4, 32);
  v += __shfl_xor(v, 8, 32);
  return v;
}

// ---------------------------------------------------------------------------
// fp32 -> bf16 conversion (weights)
// ---------------------------------------------------------------------------
__global__ __launch_bounds__(256) void cvt_kernel(const float* __restrict__ in,
                                                  bf16_t* __restrict__ out, int n) {
  int i = blockIdx.x * 256 + threadIdx.x;
  if (i < n) out[i] = (bf16_t)in[i];
}

// ---------------------------------------------------------------------------
// LayerNorm (D=768) fused with bf16 downconvert. One block per token row.
// ---------------------------------------------------------------------------
__global__ __launch_bounds__(256) void ln_to_bf16_kernel(
    const float* __restrict__ src, const float* __restrict__ g,
    const float* __restrict__ b, bf16_t* __restrict__ dst) {
  __shared__ float red[256];
  const int row = blockIdx.x;
  const int tid = threadIdx.x;
  const float* xr = src + (size_t)row * D_;
  const float v0 = xr[tid], v1 = xr[tid + 256], v2 = xr[tid + 512];
  red[tid] = v0 + v1 + v2;
  __syncthreads();
  for (int o = 128; o > 0; o >>= 1) {
    if (tid < o) red[tid] += red[tid + o];
    __syncthreads();
  }
  const float mu = red[0] * (1.0f / D_);
  __syncthreads();
  const float d0 = v0 - mu, d1 = v1 - mu, d2 = v2 - mu;
  red[tid] = d0 * d0 + d1 * d1 + d2 * d2;
  __syncthreads();
  for (int o = 128; o > 0; o >>= 1) {
    if (tid < o) red[tid] += red[tid + o];
    __syncthreads();
  }
  const float rstd = rsqrtf(red[0] * (1.0f / D_) + 1e-5f);
  bf16_t* dr = dst + (size_t)row * D_;
  dr[tid]       = (bf16_t)(d0 * rstd * g[tid]       + b[tid]);
  dr[tid + 256] = (bf16_t)(d1 * rstd * g[tid + 256] + b[tid + 256]);
  dr[tid + 512] = (bf16_t)(d2 * rstd * g[tid + 512] + b[tid + 512]);
}

// ---------------------------------------------------------------------------
// Per-head QKV projection. One wave -> one (16-token, head) tile of q,k,v.
// q,k stored [B,H,S,HD]; v stored transposed [B,H,HD,S] for PV B-fragments.
// ---------------------------------------------------------------------------
__global__ __launch_bounds__(256) void qkv_kernel(
    const bf16_t* __restrict__ xln,
    const bf16_t* __restrict__ Wqb, const float* __restrict__ bq,
    const bf16_t* __restrict__ Wkb, const float* __restrict__ bk,
    const bf16_t* __restrict__ Wvb, const float* __restrict__ bv,
    bf16_t* __restrict__ q, bf16_t* __restrict__ k, bf16_t* __restrict__ vT) {
  const int lane = threadIdx.x & 31;
  const int gw   = blockIdx.x * 8 + (threadIdx.x >> 5);  // 12288 waves
  const int h    = gw % H_;
  const int t    = gw / H_;          // token tile 0..1023
  const int row0 = t * 16;           // global token row
  const int m0   = row0 & (S_ - 1);  // row within batch sequence
  const int bh   = (t >> 6) * H_ + h;

  const v8f z = {0, 0, 0, 0, 0, 0, 0, 0};
  v8f aq[4], ak[4], av[4];
#pragma unroll
  for (int i = 0; i < 4; ++i) { aq[i] = z; ak[i] = z; av[i] = z; }

#pragma unroll
  for (int kk = 0; kk < 2; ++kk) {
    v16bf A = load_a_frag(xln + (size_t)row0 * D_ + h * HD_ + kk * 32, D_, lane);
#pragma unroll
    for (int nt = 0; nt < 4; ++nt) {
      const size_t wb = (size_t)h * HD_ * HD_ + (size_t)(nt * 16) * HD_ + kk * 32;
      aq[nt] = wmma_bf16(A, load_b_frag(Wqb + wb, HD_, lane), aq[nt]);
      ak[nt] = wmma_bf16(A, load_b_frag(Wkb + wb, HD_, lane), ak[nt]);
      av[nt] = wmma_bf16(A, load_b_frag(Wvb + wb, HD_, lane), av[nt]);
    }
  }

  const int n0 = lane & 15;
  const int mo = (lane < 16) ? 0 : 8;
#pragma unroll
  for (int nt = 0; nt < 4; ++nt) {
    const int col = nt * 16 + n0;
    const float biq = bq[h * HD_ + col];
    const float bik = bk[h * HD_ + col];
    const float biv = bv[h * HD_ + col];
#pragma unroll
    for (int r = 0; r < 8; ++r) {
      const size_t srow = (size_t)bh * S_ + m0 + mo + r;
      q[srow * HD_ + col] = (bf16_t)(aq[nt][r] + biq);
      k[srow * HD_ + col] = (bf16_t)(ak[nt][r] + bik);
      vT[((size_t)bh * HD_ + col) * S_ + (m0 + mo + r)] = (bf16_t)(av[nt][r] + biv);
    }
  }
}

// ---------------------------------------------------------------------------
// Flash attention: one wave = 16 query rows of one (b,h). Streams 32 keys/iter.
// Online softmax using the C/D register layout; P re-laid out via LDS.
// Writes out = x + attn (residual) into d_out.
// ---------------------------------------------------------------------------
__global__ __launch_bounds__(256) void attn_kernel(
    const float* __restrict__ x, const bf16_t* __restrict__ q,
    const bf16_t* __restrict__ k, const bf16_t* __restrict__ vT,
    float* __restrict__ out) {
  __shared__ __align__(64) bf16_t Plds[8][16][32];
  const int lane = threadIdx.x & 31;
  const int w    = threadIdx.x >> 5;
  const int gw   = blockIdx.x * 8 + w;   // 12288 waves
  const int h    = gw % H_;
  const int t    = gw / H_;
  const int row0 = t * 16;
  const int m0   = row0 & (S_ - 1);
  const int bh   = (t >> 6) * H_ + h;

  const bf16_t* qp = q  + ((size_t)bh * S_ + m0) * HD_;
  const bf16_t* kp = k  + (size_t)bh * S_ * HD_;
  const bf16_t* vp = vT + (size_t)bh * HD_ * S_;

  const v16bf QA0 = load_a_frag(qp, HD_, lane);       // K = 0..31
  const v16bf QA1 = load_a_frag(qp + 32, HD_, lane);  // K = 32..63

  const v8f z = {0, 0, 0, 0, 0, 0, 0, 0};
  v8f O0 = z, O1 = z, O2 = z, O3 = z;
  float mrun[8], lrun[8];
#pragma unroll
  for (int r = 0; r < 8; ++r) { mrun[r] = -1e30f; lrun[r] = 0.0f; }

  for (int kt = 0; kt < S_ / 32; ++kt) {
    const int key0 = kt * 32;
    v8f s0 = z, s1 = z;  // scores for keys [key0,key0+16) and [key0+16,key0+32)
    {
      v16bf B0 = load_b_frag(kp + (size_t)key0 * HD_, HD_, lane);
      v16bf B1 = load_b_frag(kp + (size_t)(key0 + 16) * HD_, HD_, lane);
      s0 = wmma_bf16(QA0, B0, s0);
      s1 = wmma_bf16(QA0, B1, s1);
      B0 = load_b_frag(kp + (size_t)key0 * HD_ + 32, HD_, lane);
      B1 = load_b_frag(kp + (size_t)(key0 + 16) * HD_ + 32, HD_, lane);
      s0 = wmma_bf16(QA1, B0, s0);
      s1 = wmma_bf16(QA1, B1, s1);
    }
    // Online softmax. Row M = r (+8 in upper half); per-row stats are
    // replicated across the 16 lanes of each half after the reductions.
#pragma unroll
    for (int r = 0; r < 8; ++r) {
      const float a = s0[r] * 0.125f;  // 1/sqrt(64)
      const float c = s1[r] * 0.125f;
      const float mx   = half_max(fmaxf(a, c));
      const float mnew = fmaxf(mrun[r], mx);
      const float sc   = __expf(mrun[r] - mnew);
      const float p0 = __expf(a - mnew);
      const float p1 = __expf(c - mnew);
      s0[r] = p0;
      s1[r] = p1;
      lrun[r] = lrun[r] * sc + half_sum(p0 + p1);
      mrun[r] = mnew;
      O0[r] *= sc; O1[r] *= sc; O2[r] *= sc; O3[r] *= sc;
    }
    // Re-lay-out P (C/D layout -> A layout) through LDS.
    const int halfsel = lane >> 4;
    const int c0 = lane & 15;
#pragma unroll
    for (int r = 0; r < 8; ++r) {
      Plds[w][halfsel * 8 + r][c0]      = (bf16_t)s0[r];
      Plds[w][halfsel * 8 + r][16 + c0] = (bf16_t)s1[r];
    }
    asm volatile("s_wait_dscnt 0x0" ::: "memory");  // wave-internal LDS RAW
    const v16bf PA = load_a_frag(&Plds[w][0][0], 32, lane);
    O0 = wmma_bf16(PA, load_b_frag(vp + key0,            S_, lane), O0);
    O1 = wmma_bf16(PA, load_b_frag(vp + 16 * S_ + key0,  S_, lane), O1);
    O2 = wmma_bf16(PA, load_b_frag(vp + 32 * S_ + key0,  S_, lane), O2);
    O3 = wmma_bf16(PA, load_b_frag(vp + 48 * S_ + key0,  S_, lane), O3);
  }

  const int n0 = lane & 15;
  const int mo = (lane < 16) ? 0 : 8;
#pragma unroll
  for (int r = 0; r < 8; ++r) {
    const float inv = 1.0f / lrun[r];
    const size_t ro = (size_t)(row0 + mo + r) * D_ + h * HD_;
    out[ro + n0]      = x[ro + n0]      + O0[r] * inv;
    out[ro + 16 + n0] = x[ro + 16 + n0] + O1[r] * inv;
    out[ro + 32 + n0] = x[ro + 32 + n0] + O2[r] * inv;
    out[ro + 48 + n0] = x[ro + 48 + n0] + O3[r] * inv;
  }
}

// ---------------------------------------------------------------------------
// MLP GEMM core. One block = 8 waves x 32 rows each (256 M rows) sharing one
// 64-col N tile. Weights are staged in 64x128 bf16 slabs (16 KB) into LDS via
// double-buffered global_load_async_to_lds_b128 (4 per thread per stage,
// s_wait_asynccnt 0x4), giving 32 WMMAs per block barrier and 2x B-fragment
// reuse (two 16-row A tiles per wave).
// ---------------------------------------------------------------------------
template <int K>
__device__ __forceinline__ void mlp_gemm_core(
    const bf16_t* __restrict__ act,   // [M x K] row-major activations
    const bf16_t* __restrict__ Wb,    // [N x K] row-major weights
    int nt64, int row0, int tid, int lane,
    bf16_t (*Wt)[64][128], v8f acc0[4], v8f acc1[4]) {
  // this thread's 64-byte slice of the 64x128 slab: row = tid/4, 32-col quarter
  const int trow = tid >> 2;
  const int tq   = (tid & 3) * 32;
  const unsigned ldsoff0 = (unsigned)(uintptr_t)(&Wt[0][trow][tq]);
  const unsigned ldsoff1 = (unsigned)(uintptr_t)(&Wt[1][trow][tq]);
  const bf16_t* wsrc = Wb + (size_t)(nt64 * 64 + trow) * K + tq;

#pragma unroll
  for (int j = 0; j < 4; ++j)  // prefetch stage 0
    async_load_b128(ldsoff0 + j * 16, wsrc + j * 8);

  const int nst = K / 128;
  for (int st = 0; st < nst; ++st) {
    const int cur = st & 1;
    if (st + 1 < nst) {
      const bf16_t* nsrc = wsrc + (size_t)(st + 1) * 128;
      const unsigned noff = cur ? ldsoff0 : ldsoff1;
#pragma unroll
      for (int j = 0; j < 4; ++j)
        async_load_b128(noff + j * 16, nsrc + j * 8);
      asm volatile("s_wait_asynccnt 0x4" ::: "memory");  // current slab landed
    } else {
      asm volatile("s_wait_asynccnt 0x0" ::: "memory");
    }
    __syncthreads();  // slab visible to all waves
#pragma unroll
    for (int ks = 0; ks < 4; ++ks) {
      const int kk = st * 128 + ks * 32;
      v16bf A0 = load_a_frag(act + (size_t)row0 * K + kk, K, lane);
      v16bf A1 = load_a_frag(act + (size_t)(row0 + 16) * K + kk, K, lane);
#pragma unroll
      for (int ot = 0; ot < 4; ++ot) {
        v16bf Bf = load_b_frag(&Wt[cur][ot * 16][ks * 32], 128, lane);
        acc0[ot] = wmma_bf16(A0, Bf, acc0[ot]);
        acc1[ot] = wmma_bf16(A1, Bf, acc1[ot]);
      }
    }
    __syncthreads();  // all waves done reading before buffer reuse
  }
}

__global__ __launch_bounds__(256) void mlp1_kernel(
    const bf16_t* __restrict__ xln2, const bf16_t* __restrict__ W1b,
    const float* __restrict__ b1, bf16_t* __restrict__ h1) {
  __shared__ __align__(64) bf16_t Wt[2][64][128];  // 2 x 16KB double buffer
  const int tid  = threadIdx.x;
  const int lane = tid & 31;
  const int w    = tid >> 5;
  const int nt64 = blockIdx.x % (MLP_ / 64);
  const int mt8  = blockIdx.x / (MLP_ / 64);
  const int row0 = (mt8 * 8 + w) * 32;

  const v8f z = {0, 0, 0, 0, 0, 0, 0, 0};
  v8f acc0[4] = {z, z, z, z}, acc1[4] = {z, z, z, z};
  mlp_gemm_core<D_>(xln2, W1b, nt64, row0, tid, lane, Wt, acc0, acc1);

  const int n0 = lane & 15;
  const int mo = (lane < 16) ? 0 : 8;
#pragma unroll
  for (int ot = 0; ot < 4; ++ot) {
    const int col = nt64 * 64 + ot * 16 + n0;
    const float bias = b1[col];
#pragma unroll
    for (int r = 0; r < 8; ++r) {
      const float va = acc0[ot][r] + bias;
      const float vb = acc1[ot][r] + bias;
      const float ga = 0.5f * va * (1.0f + erff(va * 0.70710678118654752f));
      const float gb = 0.5f * vb * (1.0f + erff(vb * 0.70710678118654752f));
      h1[(size_t)(row0 + mo + r) * MLP_ + col]      = (bf16_t)ga;
      h1[(size_t)(row0 + 16 + mo + r) * MLP_ + col] = (bf16_t)gb;
    }
  }
}

__global__ __launch_bounds__(256) void mlp2_kernel(
    const bf16_t* __restrict__ h1, const bf16_t* __restrict__ W2b,
    const float* __restrict__ b2, float* __restrict__ out) {
  __shared__ __align__(64) bf16_t Wt[2][64][128];
  const int tid  = threadIdx.x;
  const int lane = tid & 31;
  const int w    = tid >> 5;
  const int nt64 = blockIdx.x % (D_ / 64);
  const int mt8  = blockIdx.x / (D_ / 64);
  const int row0 = (mt8 * 8 + w) * 32;

  const v8f z = {0, 0, 0, 0, 0, 0, 0, 0};
  v8f acc0[4] = {z, z, z, z}, acc1[4] = {z, z, z, z};
  mlp_gemm_core<MLP_>(h1, W2b, nt64, row0, tid, lane, Wt, acc0, acc1);

  const int n0 = lane & 15;
  const int mo = (lane < 16) ? 0 : 8;
#pragma unroll
  for (int ot = 0; ot < 4; ++ot) {
    const int col = nt64 * 64 + ot * 16 + n0;
    const float bias = b2[col];
#pragma unroll
    for (int r = 0; r < 8; ++r) {
      const size_t ia = (size_t)(row0 + mo + r) * D_ + col;
      const size_t ib = (size_t)(row0 + 16 + mo + r) * D_ + col;
      out[ia] += acc0[ot][r] + bias;
      out[ib] += acc1[ot][r] + bias;
    }
  }
}

// ---------------------------------------------------------------------------
// Host launcher
// ---------------------------------------------------------------------------
extern "C" void kernel_launch(void* const* d_in, const int* in_sizes, int n_in,
                              void* d_out, int out_size, void* d_ws, size_t ws_size,
                              hipStream_t stream) {
  (void)in_sizes; (void)n_in; (void)out_size; (void)ws_size;
  const float* x    = (const float*)d_in[0];
  const float* ln1g = (const float*)d_in[1];
  const float* ln1b = (const float*)d_in[2];
  const float* Wq   = (const float*)d_in[3];
  const float* bq   = (const float*)d_in[4];
  const float* Wk   = (const float*)d_in[5];
  const float* bk   = (const float*)d_in[6];
  const float* Wv   = (const float*)d_in[7];
  const float* bv   = (const float*)d_in[8];
  const float* ln2g = (const float*)d_in[9];
  const float* ln2b = (const float*)d_in[10];
  const float* W1   = (const float*)d_in[11];
  const float* b1   = (const float*)d_in[12];
  const float* W2   = (const float*)d_in[13];
  const float* b2   = (const float*)d_in[14];
  float* out = (float*)d_out;

  char* ws = (char*)d_ws;
  size_t off = 0;
  auto take = [&](size_t bytes) -> char* {
    char* p = ws + off;
    off = (off + bytes + 255) & ~(size_t)255;
    return p;
  };
  bf16_t* xln  = (bf16_t*)take((size_t)NT_ * D_ * 2);
  bf16_t* qb   = (bf16_t*)take((size_t)NT_ * D_ * 2);
  bf16_t* kb_  = (bf16_t*)take((size_t)NT_ * D_ * 2);
  bf16_t* vT   = (bf16_t*)take((size_t)NT_ * D_ * 2);
  bf16_t* xln2 = (bf16_t*)take((size_t)NT_ * D_ * 2);
  bf16_t* h1   = (bf16_t*)take((size_t)NT_ * MLP_ * 2);
  bf16_t* Wqb  = (bf16_t*)take((size_t)H_ * HD_ * HD_ * 2);
  bf16_t* Wkb  = (bf16_t*)take((size_t)H_ * HD_ * HD_ * 2);
  bf16_t* Wvb  = (bf16_t*)take((size_t)H_ * HD_ * HD_ * 2);
  bf16_t* W1b  = (bf16_t*)take((size_t)MLP_ * D_ * 2);
  bf16_t* W2b  = (bf16_t*)take((size_t)D_ * MLP_ * 2);

  const int nW  = H_ * HD_ * HD_;    // 49152
  const int nW1 = MLP_ * D_;         // 2359296
  cvt_kernel<<<(nW + 255) / 256, 256, 0, stream>>>(Wq, Wqb, nW);
  cvt_kernel<<<(nW + 255) / 256, 256, 0, stream>>>(Wk, Wkb, nW);
  cvt_kernel<<<(nW + 255) / 256, 256, 0, stream>>>(Wv, Wvb, nW);
  cvt_kernel<<<(nW1 + 255) / 256, 256, 0, stream>>>(W1, W1b, nW1);
  cvt_kernel<<<(nW1 + 255) / 256, 256, 0, stream>>>(W2, W2b, nW1);

  // LN1: x -> xln (bf16)
  ln_to_bf16_kernel<<<NT_, 256, 0, stream>>>(x, ln1g, ln1b, xln);

  // QKV: 12288 waves
  qkv_kernel<<<1536, 256, 0, stream>>>(xln, Wqb, bq, Wkb, bk, Wvb, bv, qb, kb_, vT);

  // Attention + residual -> out
  attn_kernel<<<1536, 256, 0, stream>>>(x, qb, kb_, vT, out);

  // LN2: out -> xln2 (bf16)
  ln_to_bf16_kernel<<<NT_, 256, 0, stream>>>(out, ln2g, ln2b, xln2);

  // MLP: blocks = (M-groups of 256 rows) x (N-tiles of 64)
  mlp1_kernel<<<(NT_ / 256) * (MLP_ / 64), 256, 0, stream>>>(xln2, W1b, b1, h1);
  mlp2_kernel<<<(NT_ / 256) * (D_ / 64), 256, 0, stream>>>(h1, W2b, b2, out);
}
